// CrossAttention_40252433498404
// MI455X (gfx1250) — matrix operand order repaired and checked by reading, not compiled, not run on hardware
//
#include <hip/hip_runtime.h>

// ---------------- CDNA5 wave32 WMMA types ----------------
typedef __attribute__((ext_vector_type(16))) _Float16 v16h;
typedef __attribute__((ext_vector_type(8)))  _Float16 v8h;
typedef __attribute__((ext_vector_type(8)))  float    v8f;

#define WMMA_F32_F16(a,b,c) \
  __builtin_amdgcn_wmma_f32_16x16x32_f16(false,(a),false,(b),(short)0,(c),false,false)

// Problem constants
#define BB   4
#define NN   4096
#define JJ   512
#define DIMM 1024
#define CTXD 768
#define HH   16
#define DHH  64
#define HIDD 1024

// Two contiguous 16B loads -> one v16h operand (lane half h holds k = 16h..16h+15, same order for A and B)
static __device__ __forceinline__ v16h ld16(const _Float16* p0, const _Float16* p1) {
  v8h lo = *(const v8h*)p0;
  v8h hi = *(const v8h*)p1;
  v16h r;
#pragma unroll
  for (int i = 0; i < 8; ++i) { r[i] = lo[i]; r[i + 8] = hi[i]; }
  return r;
}

// ---------------- elementwise prep kernels ----------------
__global__ __launch_bounds__(256) void cast_f32_f16(const float* __restrict__ s,
                                                    _Float16* __restrict__ d, int n) {
  int i = blockIdx.x * 256 + threadIdx.x;
  if (i < n) d[i] = (_Float16)s[i];
}

// Wt[n][k] = (f16) W[k][n]   (row-major W[K][N] -> row-major Wt[N][K])
__global__ __launch_bounds__(256) void cast_transpose(const float* __restrict__ W,
                                                      _Float16* __restrict__ Wt,
                                                      int K, int Ncols) {
  int i = blockIdx.x * 256 + threadIdx.x;
  if (i >= K * Ncols) return;
  int n = i / K, k = i - n * K;
  Wt[i] = (_Float16)W[(size_t)k * Ncols + n];
}

// full_mask = pad(mask, left=True); additive form: 0 keep, -1e30 drop
__global__ __launch_bounds__(256) void mask_prep(const unsigned char* __restrict__ mask,
                                                 float* __restrict__ madd) {
  int i = blockIdx.x * 256 + threadIdx.x;
  if (i >= BB * JJ) return;
  int b = i >> 9, j = i & (JJ - 1);
  bool keep = (j == 0) || (mask[b * (JJ - 1) + j - 1] != 0);
  madd[i] = keep ? 0.f : -1.0e30f;
}

// ---------------- gamma-only LayerNorm (one row / block, 1024 cols) ----------------
__global__ __launch_bounds__(256) void ln_f32_to_f16(const float* __restrict__ X,
                                                     const float* __restrict__ g,
                                                     _Float16* __restrict__ out) {
  __shared__ float sm[16];
  int row = blockIdx.x;
  const float* xr = X + (size_t)row * DIMM;
  float s = 0.f, ss = 0.f;
  for (int c = threadIdx.x; c < DIMM; c += 256) { float v = xr[c]; s += v; ss += v * v; }
#pragma unroll
  for (int o = 16; o >= 1; o >>= 1) { s += __shfl_xor(s, o, 32); ss += __shfl_xor(ss, o, 32); }
  int wi = threadIdx.x >> 5;
  if ((threadIdx.x & 31) == 0) { sm[wi] = s; sm[8 + wi] = ss; }
  __syncthreads();
  float ts = 0.f, tss = 0.f;
#pragma unroll
  for (int i = 0; i < 8; ++i) { ts += sm[i]; tss += sm[8 + i]; }
  float mu = ts * (1.f / DIMM);
  float rstd = rsqrtf(tss * (1.f / DIMM) - mu * mu + 1e-5f);
  for (int c = threadIdx.x; c < DIMM; c += 256)
    out[(size_t)row * DIMM + c] = (_Float16)((xr[c] - mu) * rstd * g[c]);
}

__global__ __launch_bounds__(256) void ln_f16_to_f32(const _Float16* __restrict__ X,
                                                     const float* __restrict__ g,
                                                     float* __restrict__ out) {
  __shared__ float sm[16];
  int row = blockIdx.x;
  const _Float16* xr = X + (size_t)row * DIMM;
  float s = 0.f, ss = 0.f;
  for (int c = threadIdx.x; c < DIMM; c += 256) { float v = (float)xr[c]; s += v; ss += v * v; }
#pragma unroll
  for (int o = 16; o >= 1; o >>= 1) { s += __shfl_xor(s, o, 32); ss += __shfl_xor(ss, o, 32); }
  int wi = threadIdx.x >> 5;
  if ((threadIdx.x & 31) == 0) { sm[wi] = s; sm[8 + wi] = ss; }
  __syncthreads();
  float ts = 0.f, tss = 0.f;
#pragma unroll
  for (int i = 0; i < 8; ++i) { ts += sm[i]; tss += sm[8 + i]; }
  float mu = ts * (1.f / DIMM);
  float rstd = rsqrtf(tss * (1.f / DIMM) - mu * mu + 1e-5f);
  for (int c = threadIdx.x; c < DIMM; c += 256)
    out[(size_t)row * DIMM + c] = ((float)xr[c] - mu) * rstd * g[c];
}

// ---------------- generic WMMA GEMM: C[M,N] = A[M,K] * Bt[N,K]^T (f16 out) ----------
// block = 8 waves; wave tile 16x64; block tile 64(M) x 128(N).
// Ping-pong double buffer, unrolled x2 so no register copies are needed.
// Requires K/32 to be even (holds for K = 1024, 768).
__global__ __launch_bounds__(256) void gemm_f16(const _Float16* __restrict__ A,
                                                const _Float16* __restrict__ Bt,
                                                _Float16* __restrict__ C,
                                                int K, int lda, int ldb, int ldc) {
  int lane = threadIdx.x & 31, w = threadIdx.x >> 5;
  int l15 = lane & 15, hi = lane >> 4;
  int m0 = blockIdx.y * 64 + (w & 3) * 16;
  int n0 = blockIdx.x * 128 + (w >> 2) * 64;
  const _Float16* Ar = A + (size_t)(m0 + l15) * lda;
  const _Float16* Br[4];
#pragma unroll
  for (int t = 0; t < 4; ++t) Br[t] = Bt + (size_t)(n0 + 16 * t + l15) * ldb + 16 * hi;

  v8f acc[4] = {};
  v16h a0, a1, b0[4], b1[4];

  auto load_set = [&](v16h& a, v16h* bb, int kc) {
    a = ld16(Ar + kc + 16 * hi, Ar + kc + 16 * hi + 8);
#pragma unroll
    for (int t = 0; t < 4; ++t) bb[t] = ld16(Br[t] + kc, Br[t] + kc + 8);
  };
  auto mma_set = [&](const v16h& a, const v16h* bb) {
#pragma unroll
    for (int t = 0; t < 4; ++t) acc[t] = WMMA_F32_F16(a, bb[t], acc[t]);
  };

  load_set(a0, b0, 0);
  for (int kc = 32; kc < K - 32; kc += 64) {
    load_set(a1, b1, kc);        // in flight behind mma(set0)
    mma_set(a0, b0);
    load_set(a0, b0, kc + 32);   // in flight behind mma(set1)
    mma_set(a1, b1);
  }
  load_set(a1, b1, K - 32);
  mma_set(a0, b0);
  mma_set(a1, b1);

#pragma unroll
  for (int t = 0; t < 4; ++t)
#pragma unroll
    for (int r = 0; r < 8; ++r)   // C: VGPR r, row = r + 8*hi, col = lane&15
      C[(size_t)(m0 + r + 8 * hi) * ldc + n0 + 16 * t + l15] = (_Float16)acc[t][r];
}

// ---------------- kv -> per-head K[J][64] and V^T[64][J] ----------------
__global__ __launch_bounds__(256) void rearrange_kv(const _Float16* __restrict__ kv,
                                                    _Float16* __restrict__ Kh,
                                                    _Float16* __restrict__ Vt) {
  int i = blockIdx.x * 256 + threadIdx.x;               // B*J*H*DH
  if (i >= BB * JJ * HIDD) return;
  int d = i & 63, h = (i >> 6) & 15, j = (i >> 10) & (JJ - 1), b = i >> 19;
  size_t row = ((size_t)b * JJ + j) * (2 * HIDD);
  int bh = b * HH + h;
  Kh[((size_t)bh * JJ + j) * DHH + d] = kv[row + h * DHH + d];
  Vt[((size_t)bh * DHH + d) * JJ + j] = kv[row + HIDD + h * DHH + d];
}

// ---------------- fused flash attention: per wave 16 queries x one head ----------------
// J swept in 32-column chunks; chunk operands (K, V, mask) ping-pong double
// buffered one chunk ahead (16 chunks -> even, unrolled x2, no copies).
__global__ __launch_bounds__(256) void attn_kernel(const _Float16* __restrict__ Q,
                                                   const _Float16* __restrict__ Kh,
                                                   const _Float16* __restrict__ Vt,
                                                   const float* __restrict__ madd,
                                                   _Float16* __restrict__ Out) {
  __shared__ _Float16 lds[8][16 * 32];                  // per-wave P tile (16x32 f16)
  int lane = threadIdx.x & 31, w = threadIdx.x >> 5;
  int l15 = lane & 15, hi = lane >> 4;
  int bh = blockIdx.y, b = bh >> 4, h = bh & 15;
  int q0 = blockIdx.x * 128 + w * 16;

  const _Float16* Qr = Q + ((size_t)(b * NN + q0 + l15) * HIDD) + h * DHH;
  v16h qa0 = ld16(Qr + 16 * hi,      Qr + 8 + 16 * hi);  // d = 0..31
  v16h qa1 = ld16(Qr + 32 + 16 * hi, Qr + 40 + 16 * hi);  // d = 32..63

  const _Float16* Kb = Kh + (size_t)bh * JJ * DHH;
  const _Float16* Vb = Vt + (size_t)bh * DHH * JJ;
  const float* mb = madd + b * JJ;
  _Float16* P = lds[w];

  float rmax[8], rsum[8];
#pragma unroll
  for (int r = 0; r < 8; ++r) { rmax[r] = -3.0e38f; rsum[r] = 0.f; }
  v8f o[4] = {};

  struct Chunk { v16h kb00, kb01, kb10, kb11; v16h vm[4]; float ma0, ma1; };
  Chunk c0, c1;

  auto load_chunk = [&](Chunk& c, int j0) {
    const _Float16* kp0 = Kb + (size_t)(j0 + l15) * DHH + 16 * hi;
    const _Float16* kp1 = Kb + (size_t)(j0 + 16 + l15) * DHH + 16 * hi;
    c.kb00 = ld16(kp0, kp0 + 8);
    c.kb01 = ld16(kp0 + 32, kp0 + 40);
    c.kb10 = ld16(kp1, kp1 + 8);
    c.kb11 = ld16(kp1 + 32, kp1 + 40);
#pragma unroll
    for (int t = 0; t < 4; ++t) {
      const _Float16* vp = Vb + (size_t)(16 * t + l15) * JJ + j0 + 16 * hi;
      c.vm[t] = ld16(vp, vp + 8);
    }
    c.ma0 = mb[j0 + l15];
    c.ma1 = mb[j0 + 16 + l15];
  };

  auto process = [&](const Chunk& c) {
    // ---- S = Q K^T for 32 columns ----
    v8f s0 = {}, s1 = {};
    s0 = WMMA_F32_F16(qa0, c.kb00, s0);
    s0 = WMMA_F32_F16(qa1, c.kb01, s0);
    s1 = WMMA_F32_F16(qa0, c.kb10, s1);
    s1 = WMMA_F32_F16(qa1, c.kb11, s1);

    // ---- masked online softmax (rows live per half-wave in C layout) ----
    float osc[8];
#pragma unroll
    for (int r = 0; r < 8; ++r) {         // row handled by this lane: r + 8*hi
      float a0 = s0[r] * 0.125f + c.ma0;  // SCALE = 64^-0.5 applied once to S
      float a1 = s1[r] * 0.125f + c.ma1;
      float mx = fmaxf(a0, a1);
#pragma unroll
      for (int off = 8; off >= 1; off >>= 1) mx = fmaxf(mx, __shfl_xor(mx, off, 16));
      float nm = fmaxf(rmax[r], mx);
      float sc = __expf(rmax[r] - nm);
      float p0 = __expf(a0 - nm);
      float p1 = __expf(a1 - nm);
      float cs = p0 + p1;
#pragma unroll
      for (int off = 8; off >= 1; off >>= 1) cs += __shfl_xor(cs, off, 16);
      rsum[r] = rsum[r] * sc + cs;
      rmax[r] = nm;
      osc[r] = sc;
      // stage P in C-layout -> LDS row-major [16][32]
      P[(r + 8 * hi) * 32 + l15]      = (_Float16)p0;
      P[(r + 8 * hi) * 32 + 16 + l15] = (_Float16)p1;
    }
#pragma unroll
    for (int r = 0; r < 8; ++r) {
      o[0][r] *= osc[r]; o[1][r] *= osc[r]; o[2][r] *= osc[r]; o[3][r] *= osc[r];
    }
    asm volatile("s_wait_dscnt 0" ::: "memory");  // LDS stores visible before transpose read
    // read P back in A-layout (16x32)
    const _Float16* pp = P + l15 * 32 + 16 * hi;
    v16h pa = ld16(pp, pp + 8);
    // ---- O += P V (V operands already resident) ----
#pragma unroll
    for (int t = 0; t < 4; ++t) o[t] = WMMA_F32_F16(pa, c.vm[t], o[t]);
  };

  load_chunk(c0, 0);
  for (int j0 = 32; j0 < JJ - 32; j0 += 64) {
    load_chunk(c1, j0);        // flies behind process(c0)
    process(c0);
    load_chunk(c0, j0 + 32);   // flies behind process(c1)
    process(c1);
  }
  load_chunk(c1, JJ - 32);
  process(c0);
  process(c1);

  float inv[8];
#pragma unroll
  for (int r = 0; r < 8; ++r) inv[r] = 1.f / rsum[r];
#pragma unroll
  for (int t = 0; t < 4; ++t)
#pragma unroll
    for (int r = 0; r < 8; ++r)
      Out[(size_t)(b * NN + q0 + r + 8 * hi) * HIDD + h * DHH + 16 * t + l15] =
          (_Float16)(o[t][r] * inv[r]);
}

// ---------------- launcher ----------------
extern "C" void kernel_launch(void* const* d_in, const int* in_sizes, int n_in,
                              void* d_out, int out_size, void* d_ws, size_t ws_size,
                              hipStream_t stream) {
  const float* x        = (const float*)d_in[0];
  const float* context  = (const float*)d_in[1];
  const unsigned char* mask = (const unsigned char*)d_in[2];  // jax bool: 1 byte
  const float* gamma_x  = (const float*)d_in[3];
  const float* Wq       = (const float*)d_in[4];
  const float* Wkv      = (const float*)d_in[5];
  const float* Wo       = (const float*)d_in[6];
  const float* gamma_o  = (const float*)d_in[7];
  float* outp = (float*)d_out;
  (void)in_sizes; (void)n_in; (void)out_size; (void)ws_size;

  char* ws = (char*)d_ws;
  const size_t MB = 1ull << 20;
  _Float16* xn    = (_Float16*)(ws +  0 * MB);  // 32MB; attn-out reuses after q-proj
  _Float16* qh    = (_Float16*)(ws + 32 * MB);  // 32MB; proj reuses after attention
  _Float16* kvh   = (_Float16*)(ws + 64 * MB);  //  8MB
  _Float16* ctxh  = (_Float16*)(ws + 72 * MB);  //  3MB
  _Float16* khead = (_Float16*)(ws + 76 * MB);  //  4MB
  _Float16* vth   = (_Float16*)(ws + 80 * MB);  //  4MB
  _Float16* wqt   = (_Float16*)(ws + 84 * MB);  //  2MB
  _Float16* wkvt  = (_Float16*)(ws + 86 * MB);  //  3MB
  _Float16* wot   = (_Float16*)(ws + 89 * MB);  //  2MB
  float*    maddv = (float*)   (ws + 91 * MB);  //  8KB
  _Float16* attn  = xn;                         // overlay
  _Float16* proj  = qh;                         // overlay

  dim3 blk(256);
  // weight casts (transposed) + context cast + mask
  cast_transpose<<<dim3((DIMM * HIDD + 255) / 256), blk, 0, stream>>>(Wq, wqt, DIMM, HIDD);
  cast_transpose<<<dim3((CTXD * 2 * HIDD + 255) / 256), blk, 0, stream>>>(Wkv, wkvt, CTXD, 2 * HIDD);
  cast_transpose<<<dim3((HIDD * DIMM + 255) / 256), blk, 0, stream>>>(Wo, wot, HIDD, DIMM);
  cast_f32_f16<<<dim3((BB * JJ * CTXD + 255) / 256), blk, 0, stream>>>(context, ctxh, BB * JJ * CTXD);
  mask_prep<<<dim3((BB * JJ + 255) / 256), blk, 0, stream>>>(mask, maddv);
  // LN(x) -> xn (f16)
  ln_f32_to_f16<<<dim3(BB * NN), blk, 0, stream>>>(x, gamma_x, xn);
  // q = xn @ Wq      [16384 x 1024 x 1024]
  gemm_f16<<<dim3(HIDD / 128, (BB * NN) / 64), blk, 0, stream>>>(xn, wqt, qh, DIMM, DIMM, DIMM, HIDD);
  // kv = ctx @ Wkv   [2048 x 768 x 2048]
  gemm_f16<<<dim3((2 * HIDD) / 128, (BB * JJ) / 64), blk, 0, stream>>>(ctxh, wkvt, kvh, CTXD, CTXD, CTXD, 2 * HIDD);
  // split/transpose kv per head
  rearrange_kv<<<dim3((BB * JJ * HIDD + 255) / 256), blk, 0, stream>>>(kvh, khead, vth);
  // fused softmax(QK^T/8 + mask) V
  attn_kernel<<<dim3(NN / 128, BB * HH), blk, 0, stream>>>(qh, khead, vth, maddv, attn);
  // proj = attn @ Wo
  gemm_f16<<<dim3(DIMM / 128, (BB * NN) / 64), blk, 0, stream>>>(attn, wot, proj, HIDD, HIDD, HIDD, DIMM);
  // final LN -> f32 out
  ln_f16_to_f32<<<dim3(BB * NN), blk, 0, stream>>>(proj, gamma_o, outp);
}
